// CycleNet_EPD_new_16793322128017
// MI455X (gfx1250) — matrix operand order repaired and checked by reading, not compiled, not run on hardware
//
#include <hip/hip_runtime.h>
#include <hip/hip_bf16.h>

typedef __attribute__((ext_vector_type(16))) _Float16 v16h;
typedef __attribute__((ext_vector_type(8)))  _Float16 v8h;
typedef __attribute__((ext_vector_type(8)))  float    v8f;

#define B_    128
#define BETA_ 5
#define E_    2048
#define NH_   128

// workspace layout (float element offsets)
#define WS_S      0        // [B,5,4]            2560
#define WS_MN02   2560     // [B,5]               640
#define WS_MX02   3200
#define WS_MN13   3840
#define WS_MX13   4480     // ends 5120
#define WS_C1     5120     // [B,5,128]         81920 -> 87040
#define WS_BIAS24 87040    // [128]               128 -> 87168
#define WS_W24H   87168    // fp16 [128n][128k]  8192 floats -> 95360
#define WS_GPART  95360    // [B*16][128]      262144 -> 357504 floats (~1.43MB)

// ---------------- Pass 1: per-(b,beta) E-sum + masked min/max ----------------
__global__ __launch_bounds__(256) void k_sum_minmax(const float* __restrict__ pe,
                                                    float* __restrict__ ws) {
    const int bb  = blockIdx.x;            // b*5+beta
    const int tid = threadIdx.x;
    const float4* p = (const float4*)(pe + (size_t)bb * E_ * 4);
    float s0 = 0.f, s1 = 0.f, s2 = 0.f, s3 = 0.f;
    float mn02 = INFINITY, mx02 = -INFINITY, mn13 = INFINITY, mx13 = -INFINITY;
    #pragma unroll
    for (int it = 0; it < 8; ++it) {
        float4 v = p[tid + it * 256];
        s0 += v.x; s1 += v.y; s2 += v.z; s3 += v.w;
        if (v.x != 0.f) { mn02 = fminf(mn02, v.x); mx02 = fmaxf(mx02, v.x); }
        if (v.z != 0.f) { mn02 = fminf(mn02, v.z); mx02 = fmaxf(mx02, v.z); }
        if (v.y != 0.f) { mn13 = fminf(mn13, v.y); mx13 = fmaxf(mx13, v.y); }
        if (v.w != 0.f) { mn13 = fminf(mn13, v.w); mx13 = fmaxf(mx13, v.w); }
    }
    __shared__ float sb[4][256];
    __shared__ float mb[4][256];
    sb[0][tid] = s0;  sb[1][tid] = s1;  sb[2][tid] = s2;  sb[3][tid] = s3;
    mb[0][tid] = mn02; mb[1][tid] = mx02; mb[2][tid] = mn13; mb[3][tid] = mx13;
    for (int off = 128; off > 0; off >>= 1) {
        __syncthreads();
        if (tid < off) {
            sb[0][tid] += sb[0][tid + off];
            sb[1][tid] += sb[1][tid + off];
            sb[2][tid] += sb[2][tid + off];
            sb[3][tid] += sb[3][tid + off];
            mb[0][tid] = fminf(mb[0][tid], mb[0][tid + off]);
            mb[1][tid] = fmaxf(mb[1][tid], mb[1][tid + off]);
            mb[2][tid] = fminf(mb[2][tid], mb[2][tid + off]);
            mb[3][tid] = fmaxf(mb[3][tid], mb[3][tid + off]);
        }
    }
    if (tid == 0) {
        ws[WS_S + bb * 4 + 0] = sb[0][0];
        ws[WS_S + bb * 4 + 1] = sb[1][0];
        ws[WS_S + bb * 4 + 2] = sb[2][0];
        ws[WS_S + bb * 4 + 3] = sb[3][0];
        ws[WS_MN02 + bb] = mb[0][0];
        ws[WS_MX02 + bb] = mb[1][0];
        ws[WS_MN13 + bb] = mb[2][0];
        ws[WS_MX13 + bb] = mb[3][0];
    }
}

// ---------------- Pass 2: e1 MLP + c1 = e1@e2w1[:64] + e2b1 ----------------
__global__ __launch_bounds__(128) void k_e1_c1(const float* __restrict__ e1w1, const float* __restrict__ e1b1,
                                               const float* __restrict__ e1w2, const float* __restrict__ e1b2,
                                               const float* __restrict__ e2w1, const float* __restrict__ e2b1,
                                               float* __restrict__ ws) {
    const int bb = blockIdx.x, tid = threadIdx.x;
    __shared__ float sv[4], h[64], eo[64];
    if (tid < 4) sv[tid] = ws[WS_S + bb * 4 + tid];
    __syncthreads();
    if (tid < 64) {
        float a = e1b1[tid];
        #pragma unroll
        for (int k = 0; k < 4; ++k) a = fmaf(sv[k], e1w1[k * 64 + tid], a);
        h[tid] = fmaxf(a, 0.f);
    }
    __syncthreads();
    if (tid < 64) {
        float a = e1b2[tid];
        for (int k = 0; k < 64; ++k) a = fmaf(h[k], e1w2[k * 64 + tid], a);
        eo[tid] = a;                    // last layer of e1 MLP: no ReLU
    }
    __syncthreads();
    float a = e2b1[tid];
    for (int k = 0; k < 64; ++k) a = fmaf(eo[k], e2w1[k * 128 + tid], a);
    ws[WS_C1 + bb * 128 + tid] = a;
}

// ---------------- Pass 3: W24 = e2w2 @ e4w1 (fp16, col-major), bias24 ----------------
__global__ __launch_bounds__(128) void k_w24(const float* __restrict__ e2w2, const float* __restrict__ e2b2,
                                             const float* __restrict__ e4w1, const float* __restrict__ e4b1,
                                             float* __restrict__ ws) {
    const int k = blockIdx.x, n = threadIdx.x;
    float acc = 0.f;
    for (int j = 0; j < 128; ++j) acc = fmaf(e2w2[k * 128 + j], e4w1[j * 128 + n], acc);
    _Float16* W24h = (_Float16*)(ws + WS_W24H);
    W24h[n * 128 + k] = (_Float16)acc;           // column-major for B-fragment loads
    if (k == 0) {
        float b = e4b1[n];
        for (int j = 0; j < 128; ++j) b = fmaf(5.0f * e2b2[j], e4w1[j * 128 + n], b);
        ws[WS_BIAS24 + n] = b;
    }
}

// ---------------- Pass 4: main fused token GEMM (WMMA) ----------------
// per block: 128 tokens of one batch b; H[t][n] in LDS (fp16); G += ReLU(H @ W24 + bias24)
__global__ __launch_bounds__(256) void k_main(const float* __restrict__ pe,
                                              const float* __restrict__ e2w1,
                                              float* __restrict__ ws) {
    const int blk = blockIdx.x;
    const int b = blk >> 4, eblk = blk & 15;
    const int e0 = eblk * 128;
    const int tid = threadIdx.x;
    __shared__ float4   peS[BETA_][128];
    __shared__ _Float16 Alds[128 * 136];          // row stride 136 halves (272B) to spread banks

    for (int i = tid; i < BETA_ * 128; i += 256) {
        int beta = i >> 7, t = i & 127;
        peS[beta][t] = *(const float4*)(pe + (((size_t)(b * BETA_ + beta)) * E_ + e0 + t) * 4);
    }

    const int n = tid & 127, half = tid >> 7;
    float c1r[5];
    #pragma unroll
    for (int beta = 0; beta < 5; ++beta) c1r[beta] = ws[WS_C1 + (b * 5 + beta) * 128 + n];
    const float w0 = e2w1[64 * 128 + n], w1 = e2w1[65 * 128 + n];
    const float w2 = e2w1[66 * 128 + n], w3 = e2w1[67 * 128 + n];

    const int lane = tid & 31, wave = tid >> 5;
    const int nBase = wave * 16, ln = lane & 15, hi16 = lane >> 4;
    const _Float16* W24h = (const _Float16*)(ws + WS_W24H);
    v16h bfrag[4];
    #pragma unroll
    for (int s2 = 0; s2 < 4; ++s2) {       // B: N=lane%16, contiguous K-16 run per half-wave
        const v8h* wp = (const v8h*)(W24h + (nBase + ln) * 128 + hi16 * 16 + 32 * s2);
        v8h lo = wp[0], hi = wp[1];
        bfrag[s2] = __builtin_shufflevector(lo, hi, 0,1,2,3,4,5,6,7,8,9,10,11,12,13,14,15);
    }
    const float bias24n = ws[WS_BIAS24 + nBase + ln];
    __syncthreads();

    // H phase: H[t][n] = sum_beta ReLU(c1 + rank-4 update)
    for (int t = half * 64; t < half * 64 + 64; ++t) {
        float acc = 0.f;
        #pragma unroll
        for (int beta = 0; beta < 5; ++beta) {
            float4 v = peS[beta][t];
            float x = c1r[beta];
            x = fmaf(v.x, w0, x); x = fmaf(v.y, w1, x);
            x = fmaf(v.z, w2, x); x = fmaf(v.w, w3, x);
            acc += fmaxf(x, 0.f);
        }
        Alds[t * 136 + n] = (_Float16)acc;
    }
    __syncthreads();

    // WMMA phase: each wave owns N-tile nBase..nBase+15 over all 8 M-tiles
    float gpart = 0.f;
    const int klo = hi16 * 16;             // A: M=lane%16, contiguous K run {klo..+15}
    for (int mt = 0; mt < 8; ++mt) {
        v8f c = {};
        const _Float16* arow = Alds + (mt * 16 + ln) * 136;
        #pragma unroll
        for (int s2 = 0; s2 < 4; ++s2) {
            const v8h* ap = (const v8h*)arow;
            int i0 = (klo + 32 * s2) >> 3;
            v8h lo = ap[i0], hi = ap[i0 + 1];
            v16h a = __builtin_shufflevector(lo, hi, 0,1,2,3,4,5,6,7,8,9,10,11,12,13,14,15);
            c = __builtin_amdgcn_wmma_f32_16x16x32_f16(false, a, false, bfrag[s2],
                                                       (short)0, c, false, false);
        }
        #pragma unroll
        for (int r = 0; r < 8; ++r) gpart += fmaxf(c[r] + bias24n, 0.f);
    }
    gpart += __shfl_xor(gpart, 16, 32);    // combine M-halves (same N)
    if (!hi16) ws[WS_GPART + (size_t)blk * 128 + nBase + ln] = gpart;
}

// ---------------- Pass 5: final heads per batch ----------------
__global__ __launch_bounds__(128) void k_final(const float* __restrict__ gnn,
        const float* __restrict__ e4w2, const float* __restrict__ e4b2,
        const float* __restrict__ fw1, const float* __restrict__ fb1,
        const float* __restrict__ fw2, const float* __restrict__ fb2,
        const float* __restrict__ fw3, const float* __restrict__ fb3,
        const float* __restrict__ mw1, const float* __restrict__ mb1,
        const float* __restrict__ mw2, const float* __restrict__ mb2,
        const float* __restrict__ ws, float* __restrict__ out) {
    const int b = blockIdx.x, n = threadIdx.x;
    __shared__ float G[128], cat[256], f1b[128], feat[20], red[128];
    float Gs = 0.f;
    for (int p = 0; p < 16; ++p) Gs += ws[WS_GPART + (size_t)(b * 16 + p) * 128 + n];
    G[n] = Gs;
    if (n < 5) {
        feat[n]      = ws[WS_MN02 + b * 5 + n];
        feat[5 + n]  = ws[WS_MX02 + b * 5 + n];
        feat[10 + n] = ws[WS_MN13 + b * 5 + n];
        feat[15 + n] = ws[WS_MX13 + b * 5 + n];
    }
    cat[n] = gnn[b * 128 + n];
    __syncthreads();
    float Ls = 2048.0f * e4b2[n];
    for (int k = 0; k < 128; ++k) Ls = fmaf(G[k], e4w2[k * 128 + n], Ls);
    float f1 = fb1[n];
    for (int k = 0; k < 20; ++k) f1 = fmaf(feat[k], fw1[k * 128 + n], f1);
    cat[128 + n] = Ls;
    f1b[n] = fmaxf(f1, 0.f);
    __syncthreads();
    float hm = mb1[n];
    for (int k = 0; k < 256; ++k) hm = fmaf(cat[k], mw1[k * 128 + n], hm);
    hm = fmaxf(hm, 0.f);
    float f2 = fb2[n];
    for (int k = 0; k < 128; ++k) f2 = fmaf(f1b[k], fw2[k * 128 + n], f2);
    f2 = fmaxf(f2, 0.f);
    red[n] = hm * mw2[n] + f2 * fw3[n];
    __syncthreads();
    for (int off = 64; off > 0; off >>= 1) {
        if (n < off) red[n] += red[n + off];
        __syncthreads();
    }
    if (n == 0) out[b] = red[0] + mb2[0] + fb3[0];
}

extern "C" void kernel_launch(void* const* d_in, const int* in_sizes, int n_in,
                              void* d_out, int out_size, void* d_ws, size_t ws_size,
                              hipStream_t stream) {
    const float* pad_emb = (const float*)d_in[0];
    const float* gnn_out = (const float*)d_in[1];
    const float* e1w1 = (const float*)d_in[2];
    const float* e1b1 = (const float*)d_in[3];
    const float* e1w2 = (const float*)d_in[4];
    const float* e1b2 = (const float*)d_in[5];
    const float* e2w1 = (const float*)d_in[6];
    const float* e2b1 = (const float*)d_in[7];
    const float* e2w2 = (const float*)d_in[8];
    const float* e2b2 = (const float*)d_in[9];
    const float* e4w1 = (const float*)d_in[10];
    const float* e4b1 = (const float*)d_in[11];
    const float* e4w2 = (const float*)d_in[12];
    const float* e4b2 = (const float*)d_in[13];
    const float* fw1  = (const float*)d_in[14];
    const float* fb1  = (const float*)d_in[15];
    const float* fw2  = (const float*)d_in[16];
    const float* fb2  = (const float*)d_in[17];
    const float* fw3  = (const float*)d_in[18];
    const float* fb3  = (const float*)d_in[19];
    const float* mw1  = (const float*)d_in[20];
    const float* mb1  = (const float*)d_in[21];
    const float* mw2  = (const float*)d_in[22];
    const float* mb2  = (const float*)d_in[23];
    float* out = (float*)d_out;
    float* ws  = (float*)d_ws;

    k_sum_minmax<<<B_ * BETA_, 256, 0, stream>>>(pad_emb, ws);
    k_e1_c1<<<B_ * BETA_, 128, 0, stream>>>(e1w1, e1b1, e1w2, e1b2, e2w1, e2b1, ws);
    k_w24<<<128, 128, 0, stream>>>(e2w2, e2b2, e4w1, e4b1, ws);
    k_main<<<B_ * (E_ / 128), 256, 0, stream>>>(pad_emb, e2w1, ws);
    k_final<<<B_, 128, 0, stream>>>(gnn_out, e4w2, e4b2, fw1, fb1, fw2, fb2, fw3, fb3,
                                    mw1, mb1, mw2, mb2, ws, out);
}
